// GraphPropLayer_90744069030597
// MI455X (gfx1250) — matrix operand order repaired and checked by reading, not compiled, not run on hardware
//
#include <hip/hip_runtime.h>

#define N_NODES 50000
#define N_EDGES 800000
#define D_NODE 64
#define D_MSG 128

typedef __attribute__((ext_vector_type(16))) __bf16 v16bf;
typedef __attribute__((ext_vector_type(8)))  __bf16 v8bf;
typedef __attribute__((ext_vector_type(8)))  float  v8f;
typedef __attribute__((ext_vector_type(4)))  float  v4f;

// ---------------------------------------------------------------------------
// WMMA helper: D = A(16x32 bf16) * B(32x16 bf16) + C(16x16 f32)
// ---------------------------------------------------------------------------
static __device__ inline v8f wmma_bf16(v16bf a, v16bf b, v8f c) {
  return __builtin_amdgcn_wmma_f32_16x16x32_bf16(
      /*neg_a=*/false, a, /*neg_b=*/false, b,
      /*c_mod=*/(short)0, c, /*reuse_a=*/false, /*reuse_b=*/false);
}

static __device__ inline v8bf cvt8(const float* __restrict__ p) {
  v4f a = *(const v4f*)p;
  v4f b = *(const v4f*)(p + 4);
  v8bf r;
  r[0] = (__bf16)a[0]; r[1] = (__bf16)a[1]; r[2] = (__bf16)a[2]; r[3] = (__bf16)a[3];
  r[4] = (__bf16)b[0]; r[5] = (__bf16)b[1]; r[6] = (__bf16)b[2]; r[7] = (__bf16)b[3];
  return r;
}

static __device__ inline v16bf cat16(v8bf a, v8bf b) {
  return __builtin_shufflevector(a, b, 0,1,2,3,4,5,6,7,8,9,10,11,12,13,14,15);
}

static __device__ inline v8bf ld8bf(const unsigned short* __restrict__ p) {
  return *(const v8bf*)p;
}

// A-row element k of concat(seg0[64], seg1[64], seg2[64]).
// With k a compile-time constant (fully unrolled kt loops) this folds to a
// fixed base pointer + constant offset -> branch-free global_load_b128.
static __device__ inline const float* seg3(const float* s0, const float* s1,
                                           const float* s2, int k) {
  return k < 64 ? s0 + k : (k < 128 ? s1 + (k - 64) : s2 + (k - 128));
}
// A-row element k of concat(seg01[128], seg2[64])
static __device__ inline const float* seg2(const float* s01, const float* s2, int k) {
  return k < 128 ? s01 + k : s2 + (k - 128);
}

static __device__ inline void atomic_add_f32(float* p, float v) {
  unsafeAtomicAdd(p, v);  // -> global_atomic_add_f32
}

// ---------------------------------------------------------------------------
// Weight prep: f32 W[K][N] -> bf16 WT[N][K]
// ---------------------------------------------------------------------------
__global__ void wprep_kernel(const float* __restrict__ W,
                             unsigned short* __restrict__ WT, int K, int N) {
  int i = blockIdx.x * blockDim.x + threadIdx.x;
  if (i >= K * N) return;
  int k = i / N, n = i % N;
  WT[(size_t)n * K + k] = __builtin_bit_cast(unsigned short, (__bf16)W[(size_t)k * N + n]);
}

// ---------------------------------------------------------------------------
// Edge pass: 32 edges / wave, 4 waves / block.
//   H  = relu(concat(ns[src],ns[dst],ef) @ W1 + b1)   (32 x 128)
//   M  = H @ W2 + b2                                   (32 x 128)
//   agg[dst] += M   (atomic scatter)
// ---------------------------------------------------------------------------
#define H_RS 136  // LDS row stride in halves (272B: bank-spread, 16B aligned)

__global__ void __launch_bounds__(128) edge_kernel(
    const float* __restrict__ node_states,
    const float* __restrict__ edge_features,
    const unsigned short* __restrict__ W1T,  // [128][192] bf16
    const float* __restrict__ b1,
    const unsigned short* __restrict__ W2T,  // [128][128] bf16
    const float* __restrict__ b2,
    const int* __restrict__ src_idx,
    const int* __restrict__ dst_idx,
    float* __restrict__ agg) {
  __shared__ unsigned short hstage[4][32 * H_RS];
  const int wave = threadIdx.x >> 5;
  const int lane = threadIdx.x & 31;
  const int lanelow = lane & 15;
  const int half = lane >> 4;
  const long ebase = (long)(blockIdx.x * 4 + wave) * 32;

  // A-fragment gather pointers: lane owns edge row M = lane&15 of each M-tile
  const long e0 = ebase + lanelow;
  const long e1 = e0 + 16;
  const float* s0a = node_states + (long)src_idx[e0] * D_NODE;
  const float* s0b = node_states + (long)dst_idx[e0] * D_NODE;
  const float* s0c = edge_features + e0 * D_NODE;
  const float* s1a = node_states + (long)src_idx[e1] * D_NODE;
  const float* s1b = node_states + (long)dst_idx[e1] * D_NODE;
  const float* s1c = edge_features + e1 * D_NODE;

  const v8f vzero = {0.f, 0.f, 0.f, 0.f, 0.f, 0.f, 0.f, 0.f};
  v8f acc[2][8];
#pragma unroll
  for (int mt = 0; mt < 2; ++mt)
#pragma unroll
    for (int nt = 0; nt < 8; ++nt) acc[mt][nt] = vzero;

  // ---- layer 1: K = 192 (6 k-tiles of 32), fully unrolled ----
#pragma unroll
  for (int kt = 0; kt < 6; ++kt) {
    const int k0 = kt * 32 + half * 8;  // lanes16-31 take K+8 / K+24 runs
    v16bf A0 = cat16(cvt8(seg3(s0a, s0b, s0c, k0)), cvt8(seg3(s0a, s0b, s0c, k0 + 16)));
    v16bf A1 = cat16(cvt8(seg3(s1a, s1b, s1c, k0)), cvt8(seg3(s1a, s1b, s1c, k0 + 16)));
    const int kb = kt * 32 + half * 8;  // B: lanes16-31 take K+8 / K+24 runs (same as A)
#pragma unroll
    for (int nt = 0; nt < 8; ++nt) {
      const unsigned short* wp = W1T + (size_t)(nt * 16 + lanelow) * 192 + kb;
      v16bf B = cat16(ld8bf(wp), ld8bf(wp + 16));
      acc[0][nt] = wmma_bf16(A0, B, acc[0][nt]);
      acc[1][nt] = wmma_bf16(A1, B, acc[1][nt]);
    }
  }

  // ---- bias + relu, stage H as bf16 rows in LDS (C-layout -> row-major) ----
  unsigned short* h = hstage[wave];
#pragma unroll
  for (int nt = 0; nt < 8; ++nt) {
    const float bias = b1[nt * 16 + lanelow];
#pragma unroll
    for (int mt = 0; mt < 2; ++mt) {
#pragma unroll
      for (int r = 0; r < 8; ++r) {
        float x = acc[mt][nt][r] + bias;
        x = x > 0.f ? x : 0.f;
        const int row = mt * 16 + r + half * 8;  // C layout: lanes16-31 hold M+8
        h[row * H_RS + nt * 16 + lanelow] = __builtin_bit_cast(unsigned short, (__bf16)x);
      }
    }
  }
  __syncthreads();

  // ---- layer 2: K = 128 (4 k-tiles), A re-loaded from LDS in A-layout ----
  v8f acc2[2][8];
#pragma unroll
  for (int mt = 0; mt < 2; ++mt)
#pragma unroll
    for (int nt = 0; nt < 8; ++nt) acc2[mt][nt] = vzero;

  const unsigned short* p0 = h + (size_t)lanelow * H_RS;
  const unsigned short* p1 = h + (size_t)(16 + lanelow) * H_RS;
#pragma unroll
  for (int kt = 0; kt < 4; ++kt) {
    const int k0 = kt * 32 + half * 8;
    v16bf A0 = cat16(ld8bf(p0 + k0), ld8bf(p0 + k0 + 16));
    v16bf A1 = cat16(ld8bf(p1 + k0), ld8bf(p1 + k0 + 16));
    const int kb = kt * 32 + half * 8;
#pragma unroll
    for (int nt = 0; nt < 8; ++nt) {
      const unsigned short* wp = W2T + (size_t)(nt * 16 + lanelow) * 128 + kb;
      v16bf B = cat16(ld8bf(wp), ld8bf(wp + 16));
      acc2[0][nt] = wmma_bf16(A0, B, acc2[0][nt]);
      acc2[1][nt] = wmma_bf16(A1, B, acc2[1][nt]);
    }
  }

  // ---- bias + atomic scatter-add into agg[dst] ----
  int drow[2][8];
#pragma unroll
  for (int mt = 0; mt < 2; ++mt)
#pragma unroll
    for (int r = 0; r < 8; ++r)
      drow[mt][r] = dst_idx[ebase + mt * 16 + r + half * 8];

#pragma unroll
  for (int nt = 0; nt < 8; ++nt) {
    const int col = nt * 16 + lanelow;
    const float bias = b2[col];
#pragma unroll
    for (int mt = 0; mt < 2; ++mt)
#pragma unroll
      for (int r = 0; r < 8; ++r)
        atomic_add_f32(&agg[(long)drow[mt][r] * D_MSG + col], acc2[mt][nt][r] + bias);
  }
}

// ---------------------------------------------------------------------------
// Node update: out = ns + relu(concat(agg,ns) @ Wn1 + bn1) @ Wn2 + bn2
// 32 nodes / wave, 4 waves / block.
// ---------------------------------------------------------------------------
__global__ void __launch_bounds__(128) node_kernel(
    const float* __restrict__ node_states,
    const float* __restrict__ agg,
    const unsigned short* __restrict__ Wn1T,  // [128][192]
    const float* __restrict__ bn1,
    const unsigned short* __restrict__ Wn2T,  // [64][128]
    const float* __restrict__ bn2,
    float* __restrict__ out) {
  __shared__ unsigned short hstage[4][32 * H_RS];
  const int wave = threadIdx.x >> 5;
  const int lane = threadIdx.x & 31;
  const int lanelow = lane & 15;
  const int half = lane >> 4;
  const long nbase = (long)(blockIdx.x * 4 + wave) * 32;

  long n0 = nbase + lanelow;      if (n0 >= N_NODES) n0 = N_NODES - 1;
  long n1 = nbase + 16 + lanelow; if (n1 >= N_NODES) n1 = N_NODES - 1;
  const float* g0 = agg + n0 * D_MSG;
  const float* f0 = node_states + n0 * D_NODE;
  const float* g1 = agg + n1 * D_MSG;
  const float* f1 = node_states + n1 * D_NODE;

  const v8f vzero = {0.f, 0.f, 0.f, 0.f, 0.f, 0.f, 0.f, 0.f};
  v8f acc[2][8];
#pragma unroll
  for (int mt = 0; mt < 2; ++mt)
#pragma unroll
    for (int nt = 0; nt < 8; ++nt) acc[mt][nt] = vzero;

#pragma unroll
  for (int kt = 0; kt < 6; ++kt) {
    const int k0 = kt * 32 + half * 8;
    v16bf A0 = cat16(cvt8(seg2(g0, f0, k0)), cvt8(seg2(g0, f0, k0 + 16)));
    v16bf A1 = cat16(cvt8(seg2(g1, f1, k0)), cvt8(seg2(g1, f1, k0 + 16)));
    const int kb = kt * 32 + half * 8;
#pragma unroll
    for (int nt = 0; nt < 8; ++nt) {
      const unsigned short* wp = Wn1T + (size_t)(nt * 16 + lanelow) * 192 + kb;
      v16bf B = cat16(ld8bf(wp), ld8bf(wp + 16));
      acc[0][nt] = wmma_bf16(A0, B, acc[0][nt]);
      acc[1][nt] = wmma_bf16(A1, B, acc[1][nt]);
    }
  }

  unsigned short* h = hstage[wave];
#pragma unroll
  for (int nt = 0; nt < 8; ++nt) {
    const float bias = bn1[nt * 16 + lanelow];
#pragma unroll
    for (int mt = 0; mt < 2; ++mt) {
#pragma unroll
      for (int r = 0; r < 8; ++r) {
        float x = acc[mt][nt][r] + bias;
        x = x > 0.f ? x : 0.f;
        const int row = mt * 16 + r + half * 8;
        h[row * H_RS + nt * 16 + lanelow] = __builtin_bit_cast(unsigned short, (__bf16)x);
      }
    }
  }
  __syncthreads();

  v8f acc2[2][4];
#pragma unroll
  for (int mt = 0; mt < 2; ++mt)
#pragma unroll
    for (int nt = 0; nt < 4; ++nt) acc2[mt][nt] = vzero;

  const unsigned short* p0 = h + (size_t)lanelow * H_RS;
  const unsigned short* p1 = h + (size_t)(16 + lanelow) * H_RS;
#pragma unroll
  for (int kt = 0; kt < 4; ++kt) {
    const int k0 = kt * 32 + half * 8;
    v16bf A0 = cat16(ld8bf(p0 + k0), ld8bf(p0 + k0 + 16));
    v16bf A1 = cat16(ld8bf(p1 + k0), ld8bf(p1 + k0 + 16));
    const int kb = kt * 32 + half * 8;
#pragma unroll
    for (int nt = 0; nt < 4; ++nt) {
      const unsigned short* wp = Wn2T + (size_t)(nt * 16 + lanelow) * 128 + kb;
      v16bf B = cat16(ld8bf(wp), ld8bf(wp + 16));
      acc2[0][nt] = wmma_bf16(A0, B, acc2[0][nt]);
      acc2[1][nt] = wmma_bf16(A1, B, acc2[1][nt]);
    }
  }

#pragma unroll
  for (int nt = 0; nt < 4; ++nt) {
    const int col = nt * 16 + lanelow;
    const float bias = bn2[col];
#pragma unroll
    for (int mt = 0; mt < 2; ++mt) {
#pragma unroll
      for (int r = 0; r < 8; ++r) {
        const long node = nbase + mt * 16 + r + half * 8;
        if (node < N_NODES) {
          out[node * D_NODE + col] =
              node_states[node * D_NODE + col] + acc2[mt][nt][r] + bias;
        }
      }
    }
  }
}

// ---------------------------------------------------------------------------
// Host launch
// ---------------------------------------------------------------------------
extern "C" void kernel_launch(void* const* d_in, const int* in_sizes, int n_in,
                              void* d_out, int out_size, void* d_ws, size_t ws_size,
                              hipStream_t stream) {
  (void)in_sizes; (void)n_in; (void)out_size; (void)ws_size;
  const float* node_states   = (const float*)d_in[0];
  const float* edge_features = (const float*)d_in[1];
  const float* W1_f = (const float*)d_in[2];  const float* b1_f = (const float*)d_in[3];
  const float* W2_f = (const float*)d_in[4];  const float* b2_f = (const float*)d_in[5];
  const float* W1_r = (const float*)d_in[6];  const float* b1_r = (const float*)d_in[7];
  const float* W2_r = (const float*)d_in[8];  const float* b2_r = (const float*)d_in[9];
  const float* Wn1  = (const float*)d_in[10]; const float* bn1  = (const float*)d_in[11];
  const float* Wn2  = (const float*)d_in[12]; const float* bn2  = (const float*)d_in[13];
  const int* from_idx = (const int*)d_in[14];
  const int* to_idx   = (const int*)d_in[15];
  float* out = (float*)d_out;

  // workspace layout
  char* ws = (char*)d_ws;
  float* agg = (float*)ws;                               // 50000*128 f32
  size_t off = (size_t)N_NODES * D_MSG * sizeof(float);  // 25.6 MB
  unsigned short* W1fT = (unsigned short*)(ws + off); off += (size_t)192 * 128 * 2;
  unsigned short* W2fT = (unsigned short*)(ws + off); off += (size_t)128 * 128 * 2;
  unsigned short* W1rT = (unsigned short*)(ws + off); off += (size_t)192 * 128 * 2;
  unsigned short* W2rT = (unsigned short*)(ws + off); off += (size_t)128 * 128 * 2;
  unsigned short* Wn1T = (unsigned short*)(ws + off); off += (size_t)192 * 128 * 2;
  unsigned short* Wn2T = (unsigned short*)(ws + off); off += (size_t)128 * 64 * 2;

  hipMemsetAsync(agg, 0, (size_t)N_NODES * D_MSG * sizeof(float), stream);

  auto prep = [&](const float* W, unsigned short* WT, int K, int N) {
    int total = K * N;
    wprep_kernel<<<(total + 255) / 256, 256, 0, stream>>>(W, WT, K, N);
  };
  prep(W1_f, W1fT, 192, 128);
  prep(W2_f, W2fT, 128, 128);
  prep(W1_r, W1rT, 192, 128);
  prep(W2_r, W2rT, 128, 128);
  prep(Wn1,  Wn1T, 192, 128);
  prep(Wn2,  Wn2T, 128, 64);

  // 800000 edges / (4 waves * 32 edges) = 6250 blocks, exact
  const int edge_blocks = N_EDGES / 128;
  // forward: src=from, scatter to to_idx
  edge_kernel<<<edge_blocks, 128, 0, stream>>>(
      node_states, edge_features, W1fT, b1_f, W2fT, b2_f, from_idx, to_idx, agg);
  // reverse: src=to, scatter to from_idx, separate weights
  edge_kernel<<<edge_blocks, 128, 0, stream>>>(
      node_states, edge_features, W1rT, b1_r, W2rT, b2_r, to_idx, from_idx, agg);

  const int node_blocks = (N_NODES + 127) / 128;
  node_kernel<<<node_blocks, 128, 0, stream>>>(
      node_states, agg, Wn1T, bn1, Wn2T, bn2, out);
}
